// MultiHeadAttention_21199958573181
// MI455X (gfx1250) — matrix operand order repaired and checked by reading, not compiled, not run on hardware
//
#include <hip/hip_runtime.h>

// ---------------------------------------------------------------------------
// MHA forward for MI455X (gfx1250, wave32, WMMA).
// B=2, N=2048, F=1024, H=16, D=64.  All contractions run on
// v_wmma_f32_16x16x32_f16 (f16 in, f32 accumulate).
// GEMMs: 128x128 block tile, 8 waves (4x2), 32x64 per wave, K-slice 64
//        -> 16 WMMA per wave between barriers.  Tile staging uses the CDNA5
//        ASYNCcnt path (global_load_async_to_lds_b128) when available.
// ---------------------------------------------------------------------------

typedef __attribute__((ext_vector_type(16))) _Float16 v16h;
typedef __attribute__((ext_vector_type(8)))  _Float16 v8h;
typedef __attribute__((ext_vector_type(4)))  _Float16 v4h;
typedef __attribute__((ext_vector_type(8)))  float    v8f;

#define MB_  (1u << 20)
#define BATCH 2
#define SEQ   2048
#define FDIM  1024
#define HEADS 16
#define HDIM  64
#define MROWS (BATCH * SEQ)      // 4096
#define ATT_SCALE 0.125f         // 64^-0.5

// ---- CDNA5 async global->LDS copies (ASYNCcnt path) ----
#if defined(__has_builtin)
# if __has_builtin(__builtin_amdgcn_global_load_async_to_lds_b128)
#  define HAVE_ASYNC_COPY 1
# endif
#endif
#ifndef HAVE_ASYNC_COPY
# define HAVE_ASYNC_COPY 0
#endif

#if HAVE_ASYNC_COPY
typedef int b128_t __attribute__((vector_size(16)));
typedef __attribute__((address_space(1))) b128_t* gptr_b128;  // global src
typedef __attribute__((address_space(3))) b128_t* lptr_b128;  // LDS dst
#endif

__device__ __forceinline__ void stage_b128(const _Float16* gp, _Float16* lp) {
#if HAVE_ASYNC_COPY
    __builtin_amdgcn_global_load_async_to_lds_b128(
        (gptr_b128)(void*)gp, (lptr_b128)(void*)lp, 0, 0);
#else
    *(v8h*)lp = *(const v8h*)gp;
#endif
}

__device__ __forceinline__ void stage_wait() {
#if HAVE_ASYNC_COPY
# if defined(__has_builtin) && __has_builtin(__builtin_amdgcn_s_wait_asynccnt)
    __builtin_amdgcn_s_wait_asynccnt(0);
# else
    asm volatile("s_wait_asynccnt 0x0" ::: "memory");
# endif
#endif
}

__device__ __forceinline__ v8f v8f_zero() {
    v8f z;
#pragma unroll
    for (int i = 0; i < 8; ++i) z[i] = 0.0f;
    return z;
}

__device__ __forceinline__ v16h combine16(v8h lo, v8h hi) {
    v16h r;
#pragma unroll
    for (int i = 0; i < 8; ++i) { r[i] = lo[i]; r[i + 8] = hi[i]; }
    return r;
}

__device__ __forceinline__ v8f wmma32(v16h a, v16h b, v8f c) {
    // 8 args: (neg_a, A, neg_b, B, c_mod, C, reuse_a, reuse_b)
    return __builtin_amdgcn_wmma_f32_16x16x32_f16(false, a, false, b,
                                                  (short)0, c, false, false);
}

// ---------------------------------------------------------------------------
// fp32 -> f16 conversion, 4 elements per thread
// ---------------------------------------------------------------------------
__global__ void __launch_bounds__(256) cvt_f32_to_f16(const float* __restrict__ in,
                                                      _Float16* __restrict__ out,
                                                      int n4) {
    int i = blockIdx.x * 256 + threadIdx.x;
    if (i < n4) {
        float4 v = ((const float4*)in)[i];
        v4h o;
        o[0] = (_Float16)v.x; o[1] = (_Float16)v.y;
        o[2] = (_Float16)v.z; o[3] = (_Float16)v.w;
        ((v4h*)out)[i] = o;
    }
}

// ---------------------------------------------------------------------------
// Shared GEMM mainloop:  C(128x128 block) += A(Mx1024) * W^T (W is Nx1024,
// row-major -> both operands K-contiguous).  8 waves in a 4(M) x 2(N) grid,
// each wave owns 32x64 (2x4 WMMA accumulators), K-slice 64 (2 k-steps).
// LDS rows padded to 72 halfs (144 B) so every b128 fragment read is aligned.
// ---------------------------------------------------------------------------
#define LDA 72
__device__ __forceinline__ void gemm_mainloop(const _Float16* __restrict__ A,
                                              const _Float16* __restrict__ W,
                                              _Float16* As, _Float16* Bs,
                                              int m0, int n0,
                                              v8f (&acc)[2][4]) {
    const int K = FDIM;
    const int tid  = threadIdx.x;
    const int lane = tid & 31;
    const int wv   = tid >> 5;
    const int waveM = wv >> 1, waveN = wv & 1;
    const int hf = lane >> 4, ln = lane & 15;

    for (int k0 = 0; k0 < K; k0 += 64) {
        // ---- stage A/B tiles: 128 rows x 64 halfs each; 8 b128 per thread ----
#pragma unroll
        for (int i = 0; i < 4; ++i) {
            int L   = tid + 256 * i;
            int row = L >> 3, c8 = L & 7;
            stage_b128(&A[(size_t)(m0 + row) * K + k0 + c8 * 8],
                       &As[row * LDA + c8 * 8]);
            stage_b128(&W[(size_t)(n0 + row) * K + k0 + c8 * 8],
                       &Bs[row * LDA + c8 * 8]);
        }
        if (k0 + 64 < K)  // warm L2 for next A tile (global_prefetch_b8)
            __builtin_prefetch(&A[(size_t)(m0 + (tid >> 3)) * K + k0 + 64], 0, 0);
        stage_wait();
        __syncthreads();

#pragma unroll
        for (int ks = 0; ks < 2; ++ks) {
            // A fragments (16x32): lane=row; lo K=ks*32+hf*8, hi +16
            v16h af[2], bf[4];
#pragma unroll
            for (int mi = 0; mi < 2; ++mi) {
                int r = (waveM * 32 + mi * 16 + ln) * LDA + ks * 32 + hf * 8;
                af[mi] = combine16(*(const v8h*)&As[r], *(const v8h*)&As[r + 16]);
            }
            // B fragments (32x16): lane=col; lo K=ks*32+hf*8, hi +16
#pragma unroll
            for (int ni = 0; ni < 4; ++ni) {
                int r = (waveN * 64 + ni * 16 + ln) * LDA + ks * 32 + hf * 8;
                bf[ni] = combine16(*(const v8h*)&Bs[r], *(const v8h*)&Bs[r + 16]);
            }
#pragma unroll
            for (int mi = 0; mi < 2; ++mi)
#pragma unroll
                for (int ni = 0; ni < 4; ++ni)
                    acc[mi][ni] = wmma32(af[mi], bf[ni], acc[mi][ni]);
        }
        __syncthreads();
    }
}

// ---------------------------------------------------------------------------
// Fused QKV projection: grid.z in {0,1,2} selects Wq/Wk/Wv and Q/K/V output.
// Output layout (B, H, N, D) f16 so attention loads are D-contiguous.
// ---------------------------------------------------------------------------
__global__ void __launch_bounds__(256) qkv_gemm_kernel(
    const _Float16* __restrict__ xh,
    const _Float16* __restrict__ Wqh, const _Float16* __restrict__ Wkh,
    const _Float16* __restrict__ Wvh,
    _Float16* __restrict__ Qh, _Float16* __restrict__ Kh,
    _Float16* __restrict__ Vh) {
    __shared__ __align__(16) _Float16 As[128 * LDA];
    __shared__ __align__(16) _Float16 Bs[128 * LDA];

    const _Float16* W = (blockIdx.z == 0) ? Wqh : (blockIdx.z == 1) ? Wkh : Wvh;
    _Float16* Out     = (blockIdx.z == 0) ? Qh  : (blockIdx.z == 1) ? Kh  : Vh;

    const int m0 = blockIdx.y * 128, n0 = blockIdx.x * 128;
    v8f acc[2][4];
#pragma unroll
    for (int i = 0; i < 2; ++i)
#pragma unroll
        for (int j = 0; j < 4; ++j) acc[i][j] = v8f_zero();

    gemm_mainloop(xh, W, As, Bs, m0, n0, acc);

    const int lane = threadIdx.x & 31, wv = threadIdx.x >> 5;
    const int waveM = wv >> 1, waveN = wv & 1;
    const int hf = lane >> 4, ln = lane & 15;
#pragma unroll
    for (int mi = 0; mi < 2; ++mi)
#pragma unroll
        for (int ni = 0; ni < 4; ++ni)
#pragma unroll
            for (int v = 0; v < 8; ++v) {
                int m = m0 + waveM * 32 + mi * 16 + v + hf * 8;
                int n = n0 + waveN * 64 + ni * 16 + ln;
                int b = m >> 11, i = m & (SEQ - 1);
                int h = n >> 6,  d = n & (HDIM - 1);
                Out[((((size_t)b * HEADS + h) * SEQ) + i) * HDIM + d] =
                    (_Float16)acc[mi][ni][v];
            }
}

// ---------------------------------------------------------------------------
// Flash attention: grid = (SEQ/128, B*H).  Block = 256 thr = 8 waves; wave w
// owns query rows [i0 + 16w, i0 + 16w + 16).  64-key tiles; online softmax.
// ---------------------------------------------------------------------------
#define VLD 72   // Vt/Ps row stride in halfs (144 B, 16B-aligned)
__global__ void __launch_bounds__(256) attention_kernel(
    const _Float16* __restrict__ Qh, const _Float16* __restrict__ Kh,
    const _Float16* __restrict__ Vh, _Float16* __restrict__ atth) {
    __shared__ __align__(16) _Float16 Vt[HDIM * VLD];      // V tile, [d][j]
    __shared__ __align__(16) _Float16 Ps[8 * 16 * VLD];    // per-wave P strips

    const int tid = threadIdx.x, lane = tid & 31, wv = tid >> 5;
    const int hf = lane >> 4, ln = lane & 15;
    const int bh = blockIdx.y;
    const int i0 = blockIdx.x * 128;

    const _Float16* Qb = Qh + (size_t)bh * SEQ * HDIM;
    const _Float16* Kb = Kh + (size_t)bh * SEQ * HDIM;
    const _Float16* Vb = Vh + (size_t)bh * SEQ * HDIM;

    // Q fragments for this wave's 16 rows: 2 K-steps of 32 over D=64.
    const int qrow = i0 + wv * 16 + ln;
    v16h qfrag[2];
#pragma unroll
    for (int ks = 0; ks < 2; ++ks) {
        const _Float16* qp = Qb + (size_t)qrow * HDIM + ks * 32 + hf * 8;
        qfrag[ks] = combine16(*(const v8h*)qp, *(const v8h*)(qp + 16));
    }

    float m_run[8], l_run[8];
    v8f Oacc[4];
#pragma unroll
    for (int v = 0; v < 8; ++v) { m_run[v] = -3.0e38f; l_run[v] = 0.0f; }
#pragma unroll
    for (int t = 0; t < 4; ++t) Oacc[t] = v8f_zero();

    _Float16* Pw = Ps + wv * 16 * VLD;

    for (int j0 = 0; j0 < SEQ; j0 += 64) {
        // ---- stage V tile transposed into LDS: Vt[d][j] ----
        {
            int jj = tid >> 2;            // key within tile
            int dg = (tid & 3) * 16;      // 16 d-values per thread
            const _Float16* vp = Vb + (size_t)(j0 + jj) * HDIM + dg;
            v8h v0 = *(const v8h*)vp;
            v8h v1 = *(const v8h*)(vp + 8);
#pragma unroll
            for (int e = 0; e < 8; ++e) {
                Vt[(dg + e) * VLD + jj]     = v0[e];
                Vt[(dg + 8 + e) * VLD + jj] = v1[e];
            }
        }
        __syncthreads();

        // ---- S = Q @ K^T for 4 key sub-tiles of 16 ----
        v8f Sacc[4];
#pragma unroll
        for (int t = 0; t < 4; ++t) {
            Sacc[t] = v8f_zero();
#pragma unroll
            for (int ks = 0; ks < 2; ++ks) {
                int krow = j0 + t * 16 + ln;   // B-frag: lane = key column
                const _Float16* kp = Kb + (size_t)krow * HDIM + ks * 32 + hf * 8;
                v16h kf = combine16(*(const v8h*)kp, *(const v8h*)(kp + 16));
                Sacc[t] = wmma32(qfrag[ks], kf, Sacc[t]);
            }
        }

        // ---- online softmax over the 16x64 strip ----
#pragma unroll
        for (int v = 0; v < 8; ++v) {
            float sv[4];
#pragma unroll
            for (int t = 0; t < 4; ++t) sv[t] = Sacc[t][v] * ATT_SCALE;
            float rmax = fmaxf(fmaxf(sv[0], sv[1]), fmaxf(sv[2], sv[3]));
#pragma unroll
            for (int mk = 1; mk < 16; mk <<= 1)
                rmax = fmaxf(rmax, __shfl_xor(rmax, mk, 32));
            float mnew = fmaxf(m_run[v], rmax);
            float fac  = __expf(m_run[v] - mnew);
            m_run[v]   = mnew;
            float rsum = 0.0f;
#pragma unroll
            for (int t = 0; t < 4; ++t) {
                float p = __expf(sv[t] - mnew);
                rsum += p;
                Sacc[t][v] = p;
            }
#pragma unroll
            for (int mk = 1; mk < 16; mk <<= 1)
                rsum += __shfl_xor(rsum, mk, 32);
            l_run[v] = l_run[v] * fac + rsum;
#pragma unroll
            for (int t = 0; t < 4; ++t) Oacc[t][v] *= fac;
        }

        // ---- P (C-layout) -> LDS -> A-fragments ----
#pragma unroll
        for (int t = 0; t < 4; ++t)
#pragma unroll
            for (int v = 0; v < 8; ++v)
                Pw[(v + hf * 8) * VLD + t * 16 + ln] = (_Float16)Sacc[t][v];

        v16h pfrag[2];
#pragma unroll
        for (int ks = 0; ks < 2; ++ks) {
            const _Float16* pp = Pw + ln * VLD + ks * 32 + hf * 8;
            pfrag[ks] = combine16(*(const v8h*)pp, *(const v8h*)(pp + 16));
        }

        // ---- O += P @ V ----
#pragma unroll
        for (int t = 0; t < 4; ++t)
#pragma unroll
            for (int ks = 0; ks < 2; ++ks) {
                const _Float16* vtp = Vt + (t * 16 + ln) * VLD + ks * 32 + hf * 8;
                v16h vf = combine16(*(const v8h*)vtp, *(const v8h*)(vtp + 16));
                Oacc[t] = wmma32(pfrag[ks], vf, Oacc[t]);
            }
        __syncthreads();
    }

    // ---- epilogue: normalize, write att (B, N, F) f16 ----
    const int b = bh >> 4, h = bh & (HEADS - 1);
#pragma unroll
    for (int t = 0; t < 4; ++t)
#pragma unroll
        for (int v = 0; v < 8; ++v) {
            float val = Oacc[t][v] / l_run[v];
            int row = i0 + wv * 16 + v + hf * 8;
            int d   = t * 16 + ln;
            atth[(((size_t)b * SEQ + row) * FDIM) + h * HDIM + d] = (_Float16)val;
        }
}

// ---------------------------------------------------------------------------
// Output projection: out = att @ Wo^T + bo, fp32 store.
// ---------------------------------------------------------------------------
__global__ void __launch_bounds__(256) out_gemm_kernel(
    const _Float16* __restrict__ atth, const _Float16* __restrict__ Woh,
    const float* __restrict__ bo, float* __restrict__ out) {
    __shared__ __align__(16) _Float16 As[128 * LDA];
    __shared__ __align__(16) _Float16 Bs[128 * LDA];

    const int m0 = blockIdx.y * 128, n0 = blockIdx.x * 128;
    v8f acc[2][4];
#pragma unroll
    for (int i = 0; i < 2; ++i)
#pragma unroll
        for (int j = 0; j < 4; ++j) acc[i][j] = v8f_zero();

    gemm_mainloop(atth, Woh, As, Bs, m0, n0, acc);

    const int lane = threadIdx.x & 31, wv = threadIdx.x >> 5;
    const int waveM = wv >> 1, waveN = wv & 1;
    const int hf = lane >> 4, ln = lane & 15;
#pragma unroll
    for (int mi = 0; mi < 2; ++mi)
#pragma unroll
        for (int ni = 0; ni < 4; ++ni)
#pragma unroll
            for (int v = 0; v < 8; ++v) {
                int m = m0 + waveM * 32 + mi * 16 + v + hf * 8;
                int n = n0 + waveN * 64 + ni * 16 + ln;
                out[(size_t)m * FDIM + n] = acc[mi][ni][v] + bo[n];
            }
}

// ---------------------------------------------------------------------------
// Host launcher
// ---------------------------------------------------------------------------
extern "C" void kernel_launch(void* const* d_in, const int* in_sizes, int n_in,
                              void* d_out, int out_size, void* d_ws, size_t ws_size,
                              hipStream_t stream) {
    const float* x  = (const float*)d_in[0];
    const float* Wq = (const float*)d_in[1];
    const float* Wk = (const float*)d_in[2];
    const float* Wv = (const float*)d_in[3];
    const float* Wo = (const float*)d_in[4];
    const float* bo = (const float*)d_in[5];
    float* out = (float*)d_out;

    char* ws = (char*)d_ws;
    _Float16* xh   = (_Float16*)(ws + 0 * MB_);   //  8 MB (4096x1024 f16)
    _Float16* Wqh  = (_Float16*)(ws + 8 * MB_);   //  2 MB each
    _Float16* Wkh  = (_Float16*)(ws + 10 * MB_);
    _Float16* Wvh  = (_Float16*)(ws + 12 * MB_);
    _Float16* Woh  = (_Float16*)(ws + 14 * MB_);
    _Float16* Qh   = (_Float16*)(ws + 16 * MB_);  //  8 MB (B,H,N,D)
    _Float16* Kh   = (_Float16*)(ws + 24 * MB_);
    _Float16* Vh   = (_Float16*)(ws + 32 * MB_);
    _Float16* atth = (_Float16*)(ws + 40 * MB_);  //  8 MB (B,N,F)

    // fp32 -> f16 conversions
    cvt_f32_to_f16<<<(MROWS * FDIM / 4 + 255) / 256, 256, 0, stream>>>(
        x, xh, MROWS * FDIM / 4);
    cvt_f32_to_f16<<<(FDIM * FDIM / 4 + 255) / 256, 256, 0, stream>>>(
        Wq, Wqh, FDIM * FDIM / 4);
    cvt_f32_to_f16<<<(FDIM * FDIM / 4 + 255) / 256, 256, 0, stream>>>(
        Wk, Wkh, FDIM * FDIM / 4);
    cvt_f32_to_f16<<<(FDIM * FDIM / 4 + 255) / 256, 256, 0, stream>>>(
        Wv, Wvh, FDIM * FDIM / 4);
    cvt_f32_to_f16<<<(FDIM * FDIM / 4 + 255) / 256, 256, 0, stream>>>(
        Wo, Woh, FDIM * FDIM / 4);

    // Q/K/V projections (z selects the weight)
    dim3 gq(FDIM / 128, MROWS / 128, 3);
    qkv_gemm_kernel<<<gq, 256, 0, stream>>>(xh, Wqh, Wkh, Wvh, Qh, Kh, Vh);

    // flash attention
    dim3 ga(SEQ / 128, BATCH * HEADS, 1);
    attention_kernel<<<ga, 256, 0, stream>>>(Qh, Kh, Vh, atth);

    // output projection + bias
    dim3 go(FDIM / 128, MROWS / 128, 1);
    out_gemm_kernel<<<go, 256, 0, stream>>>(atth, Woh, bo, out);

    (void)in_sizes; (void)n_in; (void)out_size; (void)ws_size;
}
